// MultiHeadAttention_86423331930281
// MI455X (gfx1250) — compile-verified, matrix operand order checked by reading, not run on hardware
//
#include <hip/hip_runtime.h>

typedef unsigned short u16;
typedef unsigned int   u32;
typedef __attribute__((ext_vector_type(16))) __bf16 v16bf;
typedef __attribute__((ext_vector_type(8)))  float  v8f;
typedef __attribute__((ext_vector_type(4)))  float  v4f;
typedef __attribute__((ext_vector_type(4)))  u32    v4u;
typedef __attribute__((ext_vector_type(4)))  int    v4i;

#define QKV  1024
#define EMB  64
#define NH   16
#define OUTD 1024
#define BATCH 8
#define SEQ  1024
#define SCALE 0.03125f   // 1/sqrt(1024)

// ---- helpers -------------------------------------------------------------

__device__ inline u16 f2bf(float f) {
    u32 u = __float_as_uint(f);
    u += 0x7FFFu + ((u >> 16) & 1u);   // round-to-nearest-even
    return (u16)(u >> 16);
}

struct Frag128x2 { v4u a, b; };

// WMMA 16-bit A/B fragment: lane holds row (lane&15), K blocks at kb and kb+16
__device__ inline v16bf frag_ld(const u16* p, int kb) {
    Frag128x2 f;
    f.a = *(const v4u*)(p + kb);
    f.b = *(const v4u*)(p + kb + 16);
    return __builtin_bit_cast(v16bf, f);
}

__device__ inline v8f wmma_bf16(v16bf a, v16bf b, v8f c) {
    return __builtin_amdgcn_wmma_f32_16x16x32_bf16(
        false, a, false, b, (short)0, c, false, false);
}

__device__ inline v8f v8f_zero() {
    v8f z;
#pragma unroll
    for (int i = 0; i < 8; ++i) z[i] = 0.0f;
    return z;
}

// CDNA5 async global->LDS copy (ASYNCcnt-tracked); fallback = ld+ds_store.
__device__ inline void async_ld_b128(const u16* __restrict__ g, u16* l) {
#if __has_builtin(__builtin_amdgcn_global_load_async_to_lds_b128)
    __builtin_amdgcn_global_load_async_to_lds_b128(
        (__attribute__((address_space(1))) v4i*)g,
        (__attribute__((address_space(3))) v4i*)l, 0, 0);
#else
    *(v4u*)l = *(const v4u*)g;
#endif
}

__device__ inline void wait_async0() {
#if __has_builtin(__builtin_amdgcn_s_wait_asynccnt)
    __builtin_amdgcn_s_wait_asynccnt(0);
#else
    asm volatile("s_wait_asynccnt 0" ::: "memory");
#endif
}

__device__ inline void wait_ds0() {
    asm volatile("s_wait_dscnt 0" ::: "memory");
}

// ---- weight prep ---------------------------------------------------------

__global__ void k_prep_wt(const float* __restrict__ W, u16* __restrict__ Wt) {
    int idx = blockIdx.x * 256 + threadIdx.x;          // NH*EMB*QKV = 1M
    if (idx >= NH * EMB * QKV) return;
    int d = idx % QKV;
    int r = idx / QKV;
    int e = r % EMB;
    int h = r / EMB;
    Wt[idx] = f2bf(W[((size_t)h * QKV + d) * EMB + e]);
}

__global__ void k_prep_wo(const float* __restrict__ W, u16* __restrict__ Wt) {
    int idx = blockIdx.x * 256 + threadIdx.x;          // 1M
    if (idx >= OUTD * NH * EMB) return;
    Wt[idx] = f2bf(W[idx]);
}

// ---- QKV projection ------------------------------------------------------
// VT=0: Out[S][E] row-major (Q,K).  VT=1: Out[E][S] (V transposed).

template <int VT>
__global__ __launch_bounds__(256) void k_proj(const float* __restrict__ X,
                                              const u16*  __restrict__ Wt,
                                              u16*        __restrict__ Out) {
    __shared__ __align__(16) u16 As[128 * 32];     // 8 KB bf16 A tile
    __shared__ __align__(16) u16 Bs[64 * 32];      // 4 KB bf16 B tile (async)
    __shared__ __align__(16) u16 Cs[8 * 1024];     // 16 KB epilogue staging

    const int t    = threadIdx.x;
    const int lane = t & 31, wave = t >> 5;
    const int bh   = blockIdx.y, b = bh >> 4, h = bh & 15;
    const int m0   = blockIdx.x * 128;
    const int nlo  = lane & 15, half = lane >> 4, kb = half * 8;

    v8f acc[4];
#pragma unroll
    for (int i = 0; i < 4; ++i) acc[i] = v8f_zero();

    const int arow = t >> 1, acol = (t & 1) * 16;
    const int brow = t >> 2, bcol = (t & 3) * 8;
    const float* aSrc = X + ((size_t)(b * SEQ + m0 + arow)) * QKV + acol;
    const u16*   bSrc = Wt + ((size_t)(h * EMB + brow)) * QKV + bcol;

    for (int kc = 0; kc < QKV; kc += 32) {
        // async-stage B (bf16 weights)
        async_ld_b128(bSrc + kc, &Bs[brow * 32 + bcol]);
        // stage A: 16 f32 -> 16 bf16 per thread
        v4f f0 = *(const v4f*)(aSrc + kc);
        v4f f1 = *(const v4f*)(aSrc + kc + 4);
        v4f f2 = *(const v4f*)(aSrc + kc + 8);
        v4f f3 = *(const v4f*)(aSrc + kc + 12);
        v4u p0, p1;
        p0[0] = (u32)f2bf(f0[0]) | ((u32)f2bf(f0[1]) << 16);
        p0[1] = (u32)f2bf(f0[2]) | ((u32)f2bf(f0[3]) << 16);
        p0[2] = (u32)f2bf(f1[0]) | ((u32)f2bf(f1[1]) << 16);
        p0[3] = (u32)f2bf(f1[2]) | ((u32)f2bf(f1[3]) << 16);
        p1[0] = (u32)f2bf(f2[0]) | ((u32)f2bf(f2[1]) << 16);
        p1[1] = (u32)f2bf(f2[2]) | ((u32)f2bf(f2[3]) << 16);
        p1[2] = (u32)f2bf(f3[0]) | ((u32)f2bf(f3[1]) << 16);
        p1[3] = (u32)f2bf(f3[2]) | ((u32)f2bf(f3[3]) << 16);
        *(v4u*)&As[arow * 32 + acol]     = p0;
        *(v4u*)&As[arow * 32 + acol + 8] = p1;
        wait_async0();
        __syncthreads();

        v16bf a = frag_ld(&As[(wave * 16 + nlo) * 32], kb);
        v16bf bf0 = frag_ld(&Bs[(0 * 16 + nlo) * 32], kb);
        v16bf bf1 = frag_ld(&Bs[(1 * 16 + nlo) * 32], kb);
        v16bf bf2 = frag_ld(&Bs[(2 * 16 + nlo) * 32], kb);
        v16bf bf3 = frag_ld(&Bs[(3 * 16 + nlo) * 32], kb);
        acc[0] = wmma_bf16(a, bf0, acc[0]);
        acc[1] = wmma_bf16(a, bf1, acc[1]);
        acc[2] = wmma_bf16(a, bf2, acc[2]);
        acc[3] = wmma_bf16(a, bf3, acc[3]);
        __syncthreads();
    }

    // epilogue through LDS for coalesced b128 stores
    u16* Ct = &Cs[wave * 1024];
    if (VT == 0) {
#pragma unroll
        for (int nt = 0; nt < 4; ++nt)
#pragma unroll
            for (int v = 0; v < 8; ++v)
                Ct[(v + 8 * half) * 64 + nt * 16 + nlo] = f2bf(acc[nt][v]);
        wait_ds0();
        const int row = lane >> 1, ch = (lane & 1) * 32;   // 32 elems = 64B
        const u16* src = &Ct[row * 64 + ch];
        u16* dst = Out + ((size_t)bh * SEQ + m0 + wave * 16 + row) * EMB + ch;
#pragma unroll
        for (int c = 0; c < 4; ++c)
            *(v4u*)(dst + c * 8) = *(const v4u*)(src + c * 8);
    } else {
#pragma unroll
        for (int nt = 0; nt < 4; ++nt)
#pragma unroll
            for (int v = 0; v < 8; ++v)
                Ct[(nt * 16 + nlo) * 16 + v + 8 * half] = f2bf(acc[nt][v]);
        wait_ds0();
#pragma unroll
        for (int r = 0; r < 2; ++r) {
            int e = lane * 2 + r;
            const u16* src = &Ct[e * 16];
            u16* dst = Out + ((size_t)bh * EMB + e) * SEQ + m0 + wave * 16;
            *(v4u*)(dst)     = *(const v4u*)(src);
            *(v4u*)(dst + 8) = *(const v4u*)(src + 8);
        }
    }
}

// ---- flash attention: one wave = 16 query rows of one (b,h) --------------

__global__ __launch_bounds__(256) void k_attn(const u16* __restrict__ Q,
                                              const u16* __restrict__ K,
                                              const u16* __restrict__ Vt,
                                              u16*       __restrict__ Cc) {
    __shared__ __align__(16) u16 Pb[8][16][32];   // per-wave P tile (bf16)
    __shared__ __align__(16) u16 Cs[8 * 1024];    // epilogue staging
    __shared__ float Fb[8][16];
    __shared__ float Lb[8][16];

    const int t    = threadIdx.x;
    const int lane = t & 31, wave = t >> 5;
    const int bh   = blockIdx.y, b = bh >> 4, h = bh & 15;
    const int s0   = (blockIdx.x * 8 + wave) * 16;
    const int nlo  = lane & 15, half = lane >> 4, kb = half * 8;

    const u16* Qp = Q + ((size_t)bh * SEQ + s0 + nlo) * EMB;
    v16bf q0 = frag_ld(Qp, kb);
    v16bf q1 = frag_ld(Qp + 32, kb);

    float mrow[8], lrow[8];
#pragma unroll
    for (int v = 0; v < 8; ++v) { mrow[v] = -1e30f; lrow[v] = 0.0f; }

    v8f accT[4];
#pragma unroll
    for (int i = 0; i < 4; ++i) accT[i] = v8f_zero();

    u16 (*P)[32] = Pb[wave];
    const u16* Kbase = K + (size_t)bh * SEQ * EMB;
    const u16* Vbase = Vt + (size_t)bh * EMB * SEQ;

    for (int j = 0; j < SEQ; j += 32) {
        const u16* Kp = Kbase + (size_t)j * EMB;
        // unconditional near-scope prefetch of next K/V tiles
        __builtin_prefetch(Kp + (size_t)(32 + nlo) * EMB, 0, 3);
        __builtin_prefetch(Vbase + (size_t)nlo * SEQ + j + 32, 0, 3);

        v16bf k0a = frag_ld(Kp + (size_t)nlo * EMB, kb);
        v16bf k0b = frag_ld(Kp + (size_t)nlo * EMB + 32, kb);
        v16bf k1a = frag_ld(Kp + (size_t)(16 + nlo) * EMB, kb);
        v16bf k1b = frag_ld(Kp + (size_t)(16 + nlo) * EMB + 32, kb);

        v8f sc0 = v8f_zero(), sc1 = v8f_zero();
        sc0 = wmma_bf16(q0, k0a, sc0);
        sc0 = wmma_bf16(q1, k0b, sc0);
        sc1 = wmma_bf16(q0, k1a, sc1);
        sc1 = wmma_bf16(q1, k1b, sc1);

        float fac[8];
#pragma unroll
        for (int v = 0; v < 8; ++v) {
            float a0 = sc0[v] * SCALE;
            float a1 = sc1[v] * SCALE;
            float mx = fmaxf(a0, a1);
            mx = fmaxf(mx, __shfl_xor(mx, 1, 32));
            mx = fmaxf(mx, __shfl_xor(mx, 2, 32));
            mx = fmaxf(mx, __shfl_xor(mx, 4, 32));
            mx = fmaxf(mx, __shfl_xor(mx, 8, 32));
            float mn = fmaxf(mrow[v], mx);
            fac[v]  = __expf(mrow[v] - mn);
            mrow[v] = mn;
            float p0 = __expf(a0 - mn);
            float p1 = __expf(a1 - mn);
            float rs = p0 + p1;
            rs += __shfl_xor(rs, 1, 32);
            rs += __shfl_xor(rs, 2, 32);
            rs += __shfl_xor(rs, 4, 32);
            rs += __shfl_xor(rs, 8, 32);
            lrow[v] = lrow[v] * fac[v] + rs;
            int qr = v + 8 * half;
            P[qr][nlo]      = f2bf(p0);
            P[qr][16 + nlo] = f2bf(p1);
        }
        if (nlo == 0) {
#pragma unroll
            for (int v = 0; v < 8; ++v) Fb[wave][v + 8 * half] = fac[v];
        }
        wait_ds0();

        float f = Fb[wave][nlo];
#pragma unroll
        for (int i = 0; i < 4; ++i)
#pragma unroll
            for (int v = 0; v < 8; ++v) accT[i][v] *= f;

        v16bf pfrag = frag_ld(&P[nlo][0], kb);
        const u16* Vp = Vbase + j;
        v16bf va0 = frag_ld(Vp + (size_t)(0 * 16 + nlo) * SEQ, kb);
        v16bf va1 = frag_ld(Vp + (size_t)(1 * 16 + nlo) * SEQ, kb);
        v16bf va2 = frag_ld(Vp + (size_t)(2 * 16 + nlo) * SEQ, kb);
        v16bf va3 = frag_ld(Vp + (size_t)(3 * 16 + nlo) * SEQ, kb);
        accT[0] = wmma_bf16(va0, pfrag, accT[0]);
        accT[1] = wmma_bf16(va1, pfrag, accT[1]);
        accT[2] = wmma_bf16(va2, pfrag, accT[2]);
        accT[3] = wmma_bf16(va3, pfrag, accT[3]);
    }

    if (nlo == 0) {
#pragma unroll
        for (int v = 0; v < 8; ++v) Lb[wave][v + 8 * half] = lrow[v];
    }
    wait_ds0();
    float linv = 1.0f / Lb[wave][nlo];

    // stage [16 q][64 e] tile in LDS, then coalesced b128 stores
    u16* Ct = &Cs[wave * 1024];
#pragma unroll
    for (int i = 0; i < 4; ++i)
#pragma unroll
        for (int v = 0; v < 8; ++v)
            Ct[nlo * 64 + i * 16 + v + 8 * half] = f2bf(accT[i][v] * linv);
    wait_ds0();
    const int row = lane >> 1, ch = (lane & 1) * 32;
    const u16* src = &Ct[row * 64 + ch];
    u16* dst = Cc + ((size_t)(b * SEQ + s0 + row)) * (NH * EMB) + h * EMB + ch;
#pragma unroll
    for (int c = 0; c < 4; ++c)
        *(v4u*)(dst + c * 8) = *(const v4u*)(src + c * 8);
}

// ---- output projection (double-buffered async staging) -------------------

__global__ __launch_bounds__(256) void k_oproj(const u16*  __restrict__ A,
                                               const u16*  __restrict__ Wb,
                                               const float* __restrict__ bias,
                                               float*      __restrict__ Out) {
    __shared__ __align__(16) u16  As[2][128 * 32];   // 16 KB
    __shared__ __align__(16) u16  Bs[2][64 * 32];    // 8 KB
    __shared__ __align__(16) float Cf[8][16 * 64];   // 32 KB epilogue

    const int t    = threadIdx.x;
    const int lane = t & 31, wave = t >> 5;
    const int m0   = blockIdx.x * 128;
    const int n0   = blockIdx.y * 64;
    const int nlo  = lane & 15, half = lane >> 4, kb = half * 8;

    v8f acc[4];
#pragma unroll
    for (int i = 0; i < 4; ++i) acc[i] = v8f_zero();

    const int arow = t >> 1, acol = (t & 1) * 16;
    const int brow = t >> 2, bcol = (t & 3) * 8;
    const u16* aSrc = A  + ((size_t)(m0 + arow)) * (NH * EMB) + acol;
    const u16* bSrc = Wb + ((size_t)(n0 + brow)) * (NH * EMB) + bcol;

    auto issue = [&](int kc) {
        int buf = (kc >> 5) & 1;
        async_ld_b128(aSrc + kc,     &As[buf][arow * 32 + acol]);
        async_ld_b128(aSrc + kc + 8, &As[buf][arow * 32 + acol + 8]);
        async_ld_b128(bSrc + kc,     &Bs[buf][brow * 32 + bcol]);
    };

    issue(0);
    for (int kc = 0; kc < NH * EMB; kc += 32) {
        wait_async0();
        __syncthreads();                 // current buffer ready for everyone
        if (kc + 32 < NH * EMB) issue(kc + 32);   // fill other buffer
        const int buf = (kc >> 5) & 1;

        v16bf a = frag_ld(&As[buf][(wave * 16 + nlo) * 32], kb);
        v16bf bf0 = frag_ld(&Bs[buf][(0 * 16 + nlo) * 32], kb);
        v16bf bf1 = frag_ld(&Bs[buf][(1 * 16 + nlo) * 32], kb);
        v16bf bf2 = frag_ld(&Bs[buf][(2 * 16 + nlo) * 32], kb);
        v16bf bf3 = frag_ld(&Bs[buf][(3 * 16 + nlo) * 32], kb);
        acc[0] = wmma_bf16(a, bf0, acc[0]);
        acc[1] = wmma_bf16(a, bf1, acc[1]);
        acc[2] = wmma_bf16(a, bf2, acc[2]);
        acc[3] = wmma_bf16(a, bf3, acc[3]);
    }

    // epilogue: LDS transpose staging, bias add, b128 f32 stores
    float* Ct = Cf[wave];
#pragma unroll
    for (int nt = 0; nt < 4; ++nt)
#pragma unroll
        for (int v = 0; v < 8; ++v)
            Ct[(v + 8 * half) * 64 + nt * 16 + nlo] = acc[nt][v];
    wait_ds0();
    const int row = lane >> 1, ch = (lane & 1) * 32;
    const int m = m0 + wave * 16 + row;
    const int n = n0 + ch;
    const float* src = &Ct[row * 64 + ch];
#pragma unroll
    for (int c = 0; c < 8; ++c) {
        v4f val = *(const v4f*)(src + c * 4);
        v4f bb  = *(const v4f*)(bias + n + c * 4);
        val += bb;
        *(v4f*)(Out + (size_t)m * OUTD + n + c * 4) = val;
    }
}

// ---- launcher ------------------------------------------------------------

extern "C" void kernel_launch(void* const* d_in, const int* in_sizes, int n_in,
                              void* d_out, int out_size, void* d_ws, size_t ws_size,
                              hipStream_t stream) {
    const float* q  = (const float*)d_in[0];
    const float* k  = (const float*)d_in[1];
    const float* v  = (const float*)d_in[2];
    const float* Wq = (const float*)d_in[3];
    const float* Wk = (const float*)d_in[4];
    const float* Wv = (const float*)d_in[5];
    const float* Wo = (const float*)d_in[6];
    const float* bo = (const float*)d_in[7];

    u16* Wqt  = (u16*)d_ws;
    u16* Wkt  = Wqt  + (1u << 20);
    u16* Wvt  = Wkt  + (1u << 20);
    u16* Wo16 = Wvt  + (1u << 20);
    u16* Qb   = Wo16 + (1u << 20);
    u16* Kb   = Qb   + (8u << 20);
    u16* Vtb  = Kb   + (8u << 20);
    u16* Cc   = Vtb  + (8u << 20);

    dim3 blk(256);
    k_prep_wt<<<4096, blk, 0, stream>>>(Wq, Wqt);
    k_prep_wt<<<4096, blk, 0, stream>>>(Wk, Wkt);
    k_prep_wt<<<4096, blk, 0, stream>>>(Wv, Wvt);
    k_prep_wo<<<4096, blk, 0, stream>>>(Wo, Wo16);

    k_proj<0><<<dim3(SEQ / 128, BATCH * NH), blk, 0, stream>>>(q, Wqt, Qb);
    k_proj<0><<<dim3(SEQ / 128, BATCH * NH), blk, 0, stream>>>(k, Wkt, Kb);
    k_proj<1><<<dim3(SEQ / 128, BATCH * NH), blk, 0, stream>>>(v, Wvt, Vtb);

    k_attn<<<dim3(SEQ / 128, BATCH * NH), blk, 0, stream>>>(Qb, Kb, Vtb, Cc);

    k_oproj<<<dim3(BATCH * SEQ / 128, OUTD / 64), blk, 0, stream>>>(
        Cc, Wo16, bo, (float*)d_out);
}
